// TinyTransformer_42649025249337
// MI455X (gfx1250) — matrix operand order repaired and checked by reading, not compiled, not run on hardware
//
#include <hip/hip_runtime.h>
#include <stdint.h>

// ---------------------------------------------------------------------------
// TinyTransformer forward for MI455X (gfx1250, wave32, WMMA, async-to-LDS).
// f16 operands (converted once), f32 accumulate via v_wmma_f32_16x16x32_f16.
// GEMM: 128x128 block tile, 8 waves, double-buffered async LDS staging
// (GLOBAL_LOAD_ASYNC_TO_LDS_B128 / ASYNCcnt), branch-free steady-state loop.
// ---------------------------------------------------------------------------

typedef __attribute__((ext_vector_type(16))) _Float16 v16h;
typedef __attribute__((ext_vector_type(8)))  float    v8f;

#define TT      1024
#define BBATCH  4
#define DD      1024
#define HH      16
#define DHEAD   64
#define FFD     4096
#define NROWS   (TT * BBATCH)   // 4096 token rows
#define LAYERS  4

#define LDA_PAD 40   // halves; 80B rows -> 16B-aligned async dests, conflict-free frags
#define LQK_PAD 72   // halves; 144B rows
#define LV_PAD  34
#define LP_PAD  34

__device__ __forceinline__ v8f wmma16(v16h a, v16h b, v8f c) {
    return __builtin_amdgcn_wmma_f32_16x16x32_f16(
        false, a, false, b, (short)0, c, false, false);
}

__device__ __forceinline__ v8f zero8() {
    v8f z;
#pragma unroll
    for (int i = 0; i < 8; ++i) z[i] = 0.0f;
    return z;
}

// Async copy of 16 bytes global -> LDS (per lane). ASYNCcnt-tracked.
__device__ __forceinline__ void async_b128(const void* gaddr, uint32_t lds_byte) {
    asm volatile("global_load_async_to_lds_b128 %0, %1, off"
                 :: "v"(lds_byte), "v"((unsigned long long)(uintptr_t)gaddr)
                 : "memory");
}
__device__ __forceinline__ void wait_async0() {
    asm volatile("s_wait_asynccnt 0" ::: "memory");
}
__device__ __forceinline__ void wait_async4() {
    asm volatile("s_wait_asynccnt 4" ::: "memory");
}
__device__ __forceinline__ uint32_t lds_addr(const void* p) {
    return (uint32_t)(uintptr_t)p;
}

// A fragment (16x32 f16) from row-major LDS tile (contiguous k order: k = 16*half + s).
__device__ __forceinline__ v16h load_afrag(const _Float16* lds, int stride) {
    int lane = threadIdx.x & 31;
    int row  = lane & 15;
    int half = lane >> 4;
    const _Float16* p = lds + row * stride;
    v16h a;
#pragma unroll
    for (int v = 0; v < 8; ++v) {
        int k = (half << 4) + 2 * v;
        a[2 * v]     = p[k];
        a[2 * v + 1] = p[k + 1];
    }
    return a;
}

// B fragment (32x16 f16) from transposed LDS tile: B[k][col] at ldsT[col*stride+k].
__device__ __forceinline__ v16h load_bfrag(const _Float16* ldsT, int stride) {
    int lane = threadIdx.x & 31;
    int col  = lane & 15;
    int half = lane >> 4;
    const _Float16* p = ldsT + col * stride + (half << 4);
    v16h b;
#pragma unroll
    for (int v = 0; v < 8; ++v) {
        b[2 * v]     = p[2 * v];
        b[2 * v + 1] = p[2 * v + 1];
    }
    return b;
}

// ---------------------------------------------------------------------------
// f32 -> f16 convert (contiguous)
// ---------------------------------------------------------------------------
__global__ __launch_bounds__(256)
void cvt_f32_f16(const float* __restrict__ in, _Float16* __restrict__ out, int n) {
    int i = (blockIdx.x * 256 + threadIdx.x) * 4;
    if (i + 3 < n) {
#pragma unroll
        for (int j = 0; j < 4; ++j) out[i + j] = (_Float16)in[i + j];
    }
}

// f32 (KxN) -> f16 transposed (NxK), LDS-tiled so both sides stay coalesced.
__global__ __launch_bounds__(256)
void cvt_transpose(const float* __restrict__ W, _Float16* __restrict__ Wt,
                   int K, int N) {
    __shared__ _Float16 tile[32 * 33];
    const int n0 = blockIdx.x * 32, k0 = blockIdx.y * 32;
    const int tx = threadIdx.x & 31, ty = threadIdx.x >> 5;
#pragma unroll
    for (int i = 0; i < 4; ++i) {
        int kk = ty + 8 * i;
        tile[tx * 33 + kk] = (_Float16)W[(size_t)(k0 + kk) * N + n0 + tx];
    }
    __syncthreads();
#pragma unroll
    for (int i = 0; i < 4; ++i) {
        int r = ty + 8 * i;
        Wt[(size_t)(n0 + r) * K + k0 + tx] = tile[r * 33 + tx];
    }
}

// ---------------------------------------------------------------------------
// GEMM: out = act(A[M,K](f16) @ Wt[N,K](f16)^T + bias), f32 and/or f16 out.
// 256 threads / 8 waves; block tile 128x128; wave sub-tile 32x64 (8 WMMAs/step);
// K-step 32; double-buffered async LDS, branch-free steady-state pipeline.
// ---------------------------------------------------------------------------
__device__ __forceinline__ void stage_tile_async(const _Float16* A, const _Float16* Wt,
                                                 int K, int m0, int n0, int k0,
                                                 _Float16* lA, _Float16* lB, int tid) {
#pragma unroll
    for (int i = 0; i < 2; ++i) {                 // A: 128 rows x 64B, 512 chunks
        int ch = tid + (i << 8);
        int r = ch >> 2, seg = ch & 3;
        async_b128(A + (size_t)(m0 + r) * K + k0 + seg * 8,
                   lds_addr(lA + r * LDA_PAD) + (uint32_t)(seg * 16));
    }
#pragma unroll
    for (int i = 0; i < 2; ++i) {                 // Wt: 128 rows x 64B
        int ch = tid + (i << 8);
        int r = ch >> 2, seg = ch & 3;
        async_b128(Wt + (size_t)(n0 + r) * K + k0 + seg * 8,
                   lds_addr(lB + r * LDA_PAD) + (uint32_t)(seg * 16));
    }
}

__global__ __launch_bounds__(256)
void gemm_bias_act(const _Float16* __restrict__ A, const _Float16* __restrict__ Wt,
                   const float* __restrict__ bias,
                   float* __restrict__ outF, _Float16* __restrict__ outH,
                   int M, int N, int K, int relu) {
    __shared__ _Float16 lA[2 * 128 * LDA_PAD];
    __shared__ _Float16 lB[2 * 128 * LDA_PAD];

    const int tid  = threadIdx.x;
    const int wave = tid >> 5;
    const int lane = tid & 31;
    const int half = lane >> 4;
    const int wr   = wave & 3;        // row group: rows 32*wr .. +31
    const int wc   = wave >> 2;       // col group: cols 64*wc .. +63
    const int n0   = blockIdx.x * 128;
    const int m0   = blockIdx.y * 128;
    const int ntiles = K >> 5;

    v8f acc[2][4];
#pragma unroll
    for (int a = 0; a < 2; ++a)
#pragma unroll
        for (int j = 0; j < 4; ++j) acc[a][j] = zero8();

    stage_tile_async(A, Wt, K, m0, n0, 0, lA, lB, tid);   // prologue: tile 0

    // branch-free steady state: issue t+1, wait tile t (in-order completion)
    for (int t = 0; t < ntiles - 1; ++t) {
        _Float16* cA = lA + (t & 1) * 128 * LDA_PAD;
        _Float16* cB = lB + (t & 1) * 128 * LDA_PAD;
        _Float16* nA = lA + ((t + 1) & 1) * 128 * LDA_PAD;
        _Float16* nB = lB + ((t + 1) & 1) * 128 * LDA_PAD;
        stage_tile_async(A, Wt, K, m0, n0, (t + 1) << 5, nA, nB, tid);
        wait_async4();
        __syncthreads();

        v16h a0 = load_afrag(cA + (wr * 32)      * LDA_PAD, LDA_PAD);
        v16h a1 = load_afrag(cA + (wr * 32 + 16) * LDA_PAD, LDA_PAD);
#pragma unroll
        for (int j = 0; j < 4; ++j) {
            v16h bf = load_bfrag(cB + (wc * 64 + j * 16) * LDA_PAD, LDA_PAD);
            acc[0][j] = wmma16(a0, bf, acc[0][j]);
            acc[1][j] = wmma16(a1, bf, acc[1][j]);
        }
        __syncthreads();
    }
    {   // peeled last tile
        int t = ntiles - 1;
        _Float16* cA = lA + (t & 1) * 128 * LDA_PAD;
        _Float16* cB = lB + (t & 1) * 128 * LDA_PAD;
        wait_async0();
        __syncthreads();
        v16h a0 = load_afrag(cA + (wr * 32)      * LDA_PAD, LDA_PAD);
        v16h a1 = load_afrag(cA + (wr * 32 + 16) * LDA_PAD, LDA_PAD);
#pragma unroll
        for (int j = 0; j < 4; ++j) {
            v16h bf = load_bfrag(cB + (wc * 64 + j * 16) * LDA_PAD, LDA_PAD);
            acc[0][j] = wmma16(a0, bf, acc[0][j]);
            acc[1][j] = wmma16(a1, bf, acc[1][j]);
        }
    }

#pragma unroll
    for (int a = 0; a < 2; ++a) {
#pragma unroll
        for (int j = 0; j < 4; ++j) {
            int col = n0 + wc * 64 + (j << 4) + (lane & 15);
            float bv = bias[col];
#pragma unroll
            for (int r = 0; r < 8; ++r) {
                int row = m0 + wr * 32 + (a << 4) + r + (half << 3);
                float val = acc[a][j][r] + bv;
                if (relu) val = fmaxf(val, 0.0f);
                if (outF) outF[(size_t)row * N + col] = val;
                if (outH) outH[(size_t)row * N + col] = (_Float16)val;
            }
        }
    }
}

// ---------------------------------------------------------------------------
// resnorm: out = ((x+fx)-mean)/(std_unbiased+eps); optional f16 copy.
// ---------------------------------------------------------------------------
__global__ __launch_bounds__(256)
void resnorm_kernel(const float* __restrict__ x, const float* __restrict__ fx,
                    float* __restrict__ out, _Float16* __restrict__ outh) {
    const int row = blockIdx.x;
    const int t   = threadIdx.x;
    const float* xr = x  + (size_t)row * DD;
    const float* fr = fx + (size_t)row * DD;

    float y[4], s = 0.f, sq = 0.f;
#pragma unroll
    for (int i = 0; i < 4; ++i) {
        float v = xr[t + 256 * i] + fr[t + 256 * i];
        y[i] = v; s += v; sq += v * v;
    }
#pragma unroll
    for (int m = 16; m; m >>= 1) {
        s  += __shfl_xor(s,  m, 32);
        sq += __shfl_xor(sq, m, 32);
    }
    __shared__ float ss[8], sqq[8];
    __shared__ float mu_s, isd_s;
    if ((t & 31) == 0) { ss[t >> 5] = s; sqq[t >> 5] = sq; }
    __syncthreads();
    if (t == 0) {
        float S = 0.f, Q = 0.f;
#pragma unroll
        for (int i = 0; i < 8; ++i) { S += ss[i]; Q += sqq[i]; }
        float mu  = S / (float)DD;
        float var = (Q - S * mu) / (float)(DD - 1);
        mu_s  = mu;
        isd_s = 1.0f / (sqrtf(var) + 1e-6f);
    }
    __syncthreads();
    float mu = mu_s, isd = isd_s;
    float* orow = out + (size_t)row * DD;
#pragma unroll
    for (int i = 0; i < 4; ++i) {
        float v = (y[i] - mu) * isd;
        orow[t + 256 * i] = v;
        if (outh) outh[(size_t)row * DD + t + 256 * i] = (_Float16)v;
    }
}

// ---------------------------------------------------------------------------
// Flash attention; reference softmax-over-queries == flash with Qf=k, Kf=q.
// Block = (b,h) x 64 i-rows; wave w owns 16 rows; j steps by 32.
// ---------------------------------------------------------------------------
__global__ __launch_bounds__(128)
void attn_kernel(const _Float16* __restrict__ q, const _Float16* __restrict__ k,
                 const _Float16* __restrict__ v,
                 const unsigned char* __restrict__ mask, float* __restrict__ out) {
    __shared__ _Float16 lq[32 * LQK_PAD];       // [col(j)][d]
    __shared__ _Float16 lv[64 * LV_PAD];        // [dv][kj] transposed
    __shared__ _Float16 lp[4][16 * LP_PAD];     // per-wave P staging
    __shared__ _Float16 lk[4 * 16 * LQK_PAD];   // per-wave K rows

    const int tid  = threadIdx.x;
    const int wave = tid >> 5;
    const int lane = tid & 31;
    const int half = lane >> 4;
    const int bh   = blockIdx.x;
    const int b    = bh >> 4;
    const int h    = bh & 15;
    const int hb   = h * DHEAD;
    const int i0   = blockIdx.y * 64 + (wave << 4);
    const float scale = 1.0f / 32.0f;           // 1/sqrt(D), D=1024
    const int ldqkv = HH * DHEAD;

    // async-stage this wave's 16 K rows (16 x 128B)
    _Float16* lkw = lk + wave * 16 * LQK_PAD;
#pragma unroll
    for (int i = 0; i < 4; ++i) {
        int ch = lane + (i << 5);
        int r = ch >> 3, seg = ch & 7;
        async_b128(k + ((size_t)(i0 + r) * BBATCH + b) * ldqkv + hb + seg * 8,
                   lds_addr(lkw + r * LQK_PAD) + (uint32_t)(seg * 16));
    }
    wait_async0();  // per-wave buffer, per-wave counter: no barrier needed
    v16h aK0 = load_afrag(lkw,      LQK_PAD);   // d = 0..31
    v16h aK1 = load_afrag(lkw + 32, LQK_PAD);   // d = 32..63

    v8f o[4];
#pragma unroll
    for (int t = 0; t < 4; ++t) o[t] = zero8();
    float mrow[8], lrow[8];
#pragma unroll
    for (int r = 0; r < 8; ++r) { mrow[r] = -1e30f; lrow[r] = 0.0f; }

    for (int j0 = 0; j0 < TT; j0 += 32) {
        __syncthreads();   // previous iteration's readers done
        // async-stage Q tile (32 x 128B)
#pragma unroll
        for (int i = 0; i < 2; ++i) {
            int ch = tid + (i << 7);
            int c = ch >> 3, seg = ch & 7;
            async_b128(q + ((size_t)(j0 + c) * BBATCH + b) * ldqkv + hb + seg * 8,
                       lds_addr(lq + c * LQK_PAD) + (uint32_t)(seg * 16));
        }
        // V tile (32 rows x 64 dv) transposed, element-wise (f16 source)
#pragma unroll
        for (int i = 0; i < 8; ++i) {
            int idx = tid + (i << 7);            // 1024 dv-pairs
            int dv2 = idx & 31, kr = idx >> 5;
            const _Float16* vp =
                v + ((size_t)(j0 + kr) * BBATCH + b) * ldqkv + hb + dv2 * 2;
            lv[(dv2 * 2) * LV_PAD + kr]     = vp[0];
            lv[(dv2 * 2 + 1) * LV_PAD + kr] = vp[1];
        }
        wait_async0();
        __syncthreads();

        v8f s0 = zero8(), s1 = zero8();
        {
            v16h b00 = load_bfrag(lq,                     LQK_PAD);
            v16h b01 = load_bfrag(lq + 32,                LQK_PAD);
            v16h b10 = load_bfrag(lq + 16 * LQK_PAD,      LQK_PAD);
            v16h b11 = load_bfrag(lq + 16 * LQK_PAD + 32, LQK_PAD);
            s0 = wmma16(aK0, b00, s0); s0 = wmma16(aK1, b01, s0);
            s1 = wmma16(aK0, b10, s1); s1 = wmma16(aK1, b11, s1);
        }

        int col0 = j0 + (lane & 15);
        bool m0ok = mask[(size_t)b * TT + col0] != 0;
        bool m1ok = mask[(size_t)b * TT + col0 + 16] != 0;

#pragma unroll
        for (int r = 0; r < 8; ++r) {
            float a0 = m0ok ? s0[r] * scale : -1e30f;
            float a1 = m1ok ? s1[r] * scale : -1e30f;
            float t = fmaxf(a0, a1);
#pragma unroll
            for (int m = 8; m; m >>= 1) t = fmaxf(t, __shfl_xor(t, m, 32));
            float newm = fmaxf(mrow[r], t);
            float corr = __expf(mrow[r] - newm);
            mrow[r] = newm;
            float p0 = __expf(a0 - newm);
            float p1 = __expf(a1 - newm);
            float ps = p0 + p1;
#pragma unroll
            for (int m = 8; m; m >>= 1) ps += __shfl_xor(ps, m, 32);
            lrow[r] = lrow[r] * corr + ps;
#pragma unroll
            for (int t4 = 0; t4 < 4; ++t4) o[t4][r] *= corr;
            int prow = r + (half << 3);
            lp[wave][prow * LP_PAD + (lane & 15)]      = (_Float16)p0;
            lp[wave][prow * LP_PAD + 16 + (lane & 15)] = (_Float16)p1;
        }

        v16h aP = load_afrag(lp[wave], LP_PAD);   // same-wave LDS ordering
#pragma unroll
        for (int t4 = 0; t4 < 4; ++t4) {
            v16h bV = load_bfrag(lv + (t4 << 4) * LV_PAD, LV_PAD);
            o[t4] = wmma16(aP, bV, o[t4]);
        }
    }

#pragma unroll
    for (int r = 0; r < 8; ++r) {
        float inv = 1.0f / lrow[r];
        int irow = i0 + r + (half << 3);
#pragma unroll
        for (int t4 = 0; t4 < 4; ++t4) {
            int col = hb + (t4 << 4) + (lane & 15);
            out[((size_t)irow * BBATCH + b) * ldqkv + col] = o[t4][r] * inv;
        }
    }
}

// ---------------------------------------------------------------------------
// Host orchestration (graph-capture safe).
// ---------------------------------------------------------------------------
extern "C" void kernel_launch(void* const* d_in, const int* in_sizes, int n_in,
                              void* d_out, int out_size, void* d_ws, size_t ws_size,
                              hipStream_t stream) {
    (void)in_sizes; (void)n_in; (void)out_size; (void)ws_size;

    const float*         x    = (const float*)d_in[0];
    const unsigned char* mask = (const unsigned char*)d_in[1];
    const float* Wk = (const float*)d_in[2];
    const float* bk = (const float*)d_in[3];
    const float* Wq = (const float*)d_in[4];
    const float* bq = (const float*)d_in[5];
    const float* Wv = (const float*)d_in[6];
    const float* bv = (const float*)d_in[7];
    const float* W1 = (const float*)d_in[8];
    const float* b1 = (const float*)d_in[9];
    const float* W2 = (const float*)d_in[10];
    const float* b2 = (const float*)d_in[11];
    float* outp = (float*)d_out;

    const size_t MD  = (size_t)NROWS * DD;    // 4096x1024
    const size_t MF  = (size_t)NROWS * FFD;   // 4096x4096
    const size_t DF  = (size_t)DD * FFD;
    const size_t DDq = (size_t)DD * DD;

    float* ws    = (float*)d_ws;
    float* ffout = ws;            // MD
    float* zbuf  = ffout + MD;    // MD
    float* ab    = zbuf  + MD;    // MD
    float* xn    = ab    + MD;    // MD

    _Float16* hb = (_Float16*)(xn + MD);
    _Float16* w1t = hb;            // FFD x DD
    _Float16* w2t = w1t + DF;      // DD x FFD
    _Float16* wqt = w2t + DF;      // DD x DD
    _Float16* wkt = wqt + DDq;
    _Float16* wvt = wkt + DDq;
    _Float16* xh  = wvt + DDq;     // MD
    _Float16* zh  = xh  + MD;      // MD
    _Float16* h1h = zh  + MD;      // MF
    _Float16* qh  = h1h + MF;      // MD
    _Float16* kh  = qh  + MD;
    _Float16* vh  = kh  + MD;
    _Float16* xnh = vh  + MD;      // MD

    dim3 gblk(256);
    dim3 gFF1(FFD / 128, NROWS / 128);
    dim3 gD  (DD  / 128, NROWS / 128);
    dim3 gAttn(BBATCH * HH, TT / 64);
    dim3 tblk(256);
    dim3 ablk(128);

    // initial x -> f16
    cvt_f32_f16<<<(int)(MD / 1024), tblk, 0, stream>>>(x, xh, (int)MD);

    const float*    curF = x;
    const _Float16* curH = xh;

    for (int l = 0; l < LAYERS; ++l) {
        // per-layer weight convert+transpose (reused buffers)
        cvt_transpose<<<dim3(FFD / 32, DD / 32), tblk, 0, stream>>>(
            W1 + (size_t)l * DF, w1t, DD, FFD);
        cvt_transpose<<<dim3(DD / 32, FFD / 32), tblk, 0, stream>>>(
            W2 + (size_t)l * DF, w2t, FFD, DD);
        cvt_transpose<<<dim3(DD / 32, DD / 32), tblk, 0, stream>>>(
            Wq + (size_t)l * DDq, wqt, DD, DD);
        cvt_transpose<<<dim3(DD / 32, DD / 32), tblk, 0, stream>>>(
            Wk + (size_t)l * DDq, wkt, DD, DD);
        cvt_transpose<<<dim3(DD / 32, DD / 32), tblk, 0, stream>>>(
            Wv + (size_t)l * DDq, wvt, DD, DD);

        // FF: h1h = relu(x@W1+b1) (f16); ffout = relu(h1@W2+b2) (f32)
        gemm_bias_act<<<gFF1, gblk, 0, stream>>>(curH, w1t, b1 + (size_t)l * FFD,
                                                 nullptr, h1h, NROWS, FFD, DD, 1);
        gemm_bias_act<<<gD, gblk, 0, stream>>>(h1h, w2t, b2 + (size_t)l * DD,
                                               ffout, nullptr, NROWS, DD, FFD, 1);
        // z = resnorm(x, ff(x))  (f32 + f16)
        resnorm_kernel<<<NROWS, tblk, 0, stream>>>(curF, ffout, zbuf, zh);

        // q/k/v projections (f16 outputs)
        gemm_bias_act<<<gD, gblk, 0, stream>>>(zh, wqt, bq + (size_t)l * DD,
                                               nullptr, qh, NROWS, DD, DD, 0);
        gemm_bias_act<<<gD, gblk, 0, stream>>>(zh, wkt, bk + (size_t)l * DD,
                                               nullptr, kh, NROWS, DD, DD, 0);
        gemm_bias_act<<<gD, gblk, 0, stream>>>(zh, wvt, bv + (size_t)l * DD,
                                               nullptr, vh, NROWS, DD, DD, 0);

        // attention
        attn_kernel<<<gAttn, ablk, 0, stream>>>(qh, kh, vh, mask, ab);

        // x_next = resnorm(z, attn(z))
        float* dst = (l == LAYERS - 1) ? outp : xn;
        resnorm_kernel<<<NROWS, tblk, 0, stream>>>(zbuf, ab, dst, xnh);
        curF = dst;
        curH = xnh;
    }
}
